// Wav2vec2Model_48249662603595
// MI455X (gfx1250) — matrix operand order repaired and checked by reading, not compiled, not run on hardware
//
#include <hip/hip_runtime.h>
#include <math.h>

// Problem constants (match reference)
#define Bz 8
#define Tz 1024
#define Hz 256
#define Nz 100
#define NEG_FILL (-1e9f)
#define COS_EPS  (1e-8f)

typedef __attribute__((ext_vector_type(16))) _Float16 v16h;
typedef __attribute__((ext_vector_type(8)))  float    v8f;

__global__ void w2v2_zero_out(float* out) { out[0] = 0.0f; }

// One block of 256 threads per row r in [0, B*T): qn[r]=||q_row||, fn[r]=||f_row||
__global__ void w2v2_row_norms(const float* __restrict__ q, const float* __restrict__ f,
                               float* __restrict__ qn, float* __restrict__ fn) {
    __shared__ float s1[256];
    __shared__ float s2[256];
    const int r = blockIdx.x, tid = threadIdx.x;
    const float a = q[(size_t)r * Hz + tid];
    const float b = f[(size_t)r * Hz + tid];
    s1[tid] = a * a;
    s2[tid] = b * b;
    __syncthreads();
    for (int off = 128; off > 0; off >>= 1) {
        if (tid < off) { s1[tid] += s1[tid + off]; s2[tid] += s2[tid + off]; }
        __syncthreads();
    }
    if (tid == 0) { qn[r] = sqrtf(s1[0]); fn[r] = sqrtf(s2[0]); }
}

// One wave32 per (b,t). 7 candidate groups of 16; 8 K-steps of 32 per group via
// v_wmma_f32_16x16x32_f16 (A = broadcast feature row, B = 16 gathered candidates).
__global__ void __launch_bounds__(32)
w2v2_contrastive(const float* __restrict__ q, const float* __restrict__ f,
                 const int* __restrict__ negidx, const unsigned char* __restrict__ mask,
                 const float* __restrict__ qnorm, const float* __restrict__ fnorm,
                 float* __restrict__ out) {
    const int bt = blockIdx.x;
    if (!mask[bt]) return;                    // uniform per wave: ce * 0 contribution
    const int b = bt / Tz, t = bt % Tz;
    const int lane = threadIdx.x;

    __shared__ float    qpos[Hz];             // positive row (for neg_is_pos compare)
    __shared__ _Float16 fh[Hz];               // feature row as f16 (A operand source)
    __shared__ float    lg[128];              // logits (112 used, padded)

    for (int i = 0; i < 8; i++) {
        const int j = lane * 8 + i;           // 32 lanes * 8 = 256
        qpos[j] = q[(size_t)bt * Hz + j];
        fh[j]   = (_Float16)f[(size_t)bt * Hz + j];
    }
    __syncthreads();

    // A fragments (this code's contiguous k order, same as B): all 16 M-rows identical == feature row.
    // lanes 0-15: elems 0..15 = K k0+0..15
    // lanes 16-31: elems 0..15 = K k0+16..31.
    v16h afrag[8];
    {
        const int hi = (lane >= 16) ? 16 : 0;
        for (int ks = 0; ks < 8; ks++) {
            const int base = ks * 32 + hi;
            for (int e = 0; e < 8; e++) {
                afrag[ks][e]     = fh[base + e];
                afrag[ks][8 + e] = fh[base + 8 + e];
            }
        }
    }

    const float fn    = fmaxf(fnorm[bt], COS_EPS);
    const int   col   = lane & 15;            // candidate column this lane feeds
    const int   khalf = (lane >> 4) * 16;     // lanes 0-15: K lo half; 16-31: K hi half

    for (int g = 0; g < 7; g++) {
        const int s = g * 16 + col;           // candidate slot 0..111 (0 = positive)
        int cidx;
        if (s == 0)       cidx = bt;
        else if (s <= Nz) cidx = negidx[b * (Tz * Nz) + t * Nz + (s - 1)];
        else              cidx = bt;          // pad slot: safe address, logit forced -inf
        const float* row = q + (size_t)cidx * Hz;

        v8f c = {0.f, 0.f, 0.f, 0.f, 0.f, 0.f, 0.f, 0.f};
        bool neq = false;                     // branchless: lane-mask OR accumulation
        for (int ks = 0; ks < 8; ks++) {
            const int k0 = ks * 32 + khalf;
            // gathered candidate chunk (global, 64B per lane)
            const float4 v0 = *(const float4*)(row + k0 + 0);
            const float4 v1 = *(const float4*)(row + k0 + 4);
            const float4 v2 = *(const float4*)(row + k0 + 8);
            const float4 v3 = *(const float4*)(row + k0 + 12);
            // positive chunk (LDS, vector ds_load_b128)
            const float4 p0 = *(const float4*)(qpos + k0 + 0);
            const float4 p1 = *(const float4*)(qpos + k0 + 4);
            const float4 p2 = *(const float4*)(qpos + k0 + 8);
            const float4 p3 = *(const float4*)(qpos + k0 + 12);

            neq = neq | (v0.x != p0.x) | (v0.y != p0.y) | (v0.z != p0.z) | (v0.w != p0.w)
                      | (v1.x != p1.x) | (v1.y != p1.y) | (v1.z != p1.z) | (v1.w != p1.w)
                      | (v2.x != p2.x) | (v2.y != p2.y) | (v2.z != p2.z) | (v2.w != p2.w)
                      | (v3.x != p3.x) | (v3.y != p3.y) | (v3.z != p3.z) | (v3.w != p3.w);

            v16h bvec;                        // B in contiguous k order (same as A): elem e = K khalf+e of column `col`
            bvec[0]  = (_Float16)v0.x; bvec[1]  = (_Float16)v0.y;
            bvec[2]  = (_Float16)v0.z; bvec[3]  = (_Float16)v0.w;
            bvec[4]  = (_Float16)v1.x; bvec[5]  = (_Float16)v1.y;
            bvec[6]  = (_Float16)v1.z; bvec[7]  = (_Float16)v1.w;
            bvec[8]  = (_Float16)v2.x; bvec[9]  = (_Float16)v2.y;
            bvec[10] = (_Float16)v2.z; bvec[11] = (_Float16)v2.w;
            bvec[12] = (_Float16)v3.x; bvec[13] = (_Float16)v3.y;
            bvec[14] = (_Float16)v3.z; bvec[15] = (_Float16)v3.w;

            c = __builtin_amdgcn_wmma_f32_16x16x32_f16(
                    /*neg_a=*/false, afrag[ks], /*neg_b=*/false, bvec,
                    /*c_mod=*/(short)0, c, /*reuse_a=*/false, /*reuse_b=*/false);
        }
        // combine the two K-halves of the equality test (lane j <-> lane j+16)
        const int  neqMine  = neq ? 1 : 0;
        const int  neqOther = __shfl_xor(neqMine, 16, 32);
        const bool fullEq   = ((neqMine | neqOther) == 0);

        const float qn    = fmaxf(qnorm[cidx], COS_EPS);
        float logit = c[0] * 10.0f / (fn * qn);  // D row M=0, N=lane (lanes 0-15); /0.1 temp
        if (s > 0 && fullEq) logit = NEG_FILL;
        if (s > Nz)          logit = -INFINITY;  // pad slots excluded from softmax
        if (lane < 16) lg[s] = logit;
    }
    __syncthreads();

    // log-softmax CE over lg[0..100], label 0
    float m = -INFINITY;
    for (int i = lane; i <= Nz; i += 32) m = fmaxf(m, lg[i]);
    for (int off = 16; off > 0; off >>= 1) m = fmaxf(m, __shfl_xor(m, off, 32));
    float ssum = 0.0f;
    for (int i = lane; i <= Nz; i += 32) ssum += __expf(lg[i] - m);
    for (int off = 16; off > 0; off >>= 1) ssum += __shfl_xor(ssum, off, 32);
    if (lane == 0) {
        const float ce = m + __logf(ssum) - lg[0];
        atomicAdd(out, ce);
    }
}

extern "C" void kernel_launch(void* const* d_in, const int* in_sizes, int n_in,
                              void* d_out, int out_size, void* d_ws, size_t ws_size,
                              hipStream_t stream) {
    const float*         q      = (const float*)d_in[0];          // quantized_features (B,T,H)
    const float*         f      = (const float*)d_in[1];          // features (B,T,H)
    const int*           negidx = (const int*)d_in[2];            // negative_indices (B,T*N)
    const unsigned char* mask   = (const unsigned char*)d_in[3];  // mask_time_indices (B,T) bool
    float* out = (float*)d_out;

    float* qn = (float*)d_ws;            // 8192 floats
    float* fn = qn + (Bz * Tz);          // 8192 floats (64 KB total scratch)

    w2v2_zero_out<<<1, 1, 0, stream>>>(out);
    w2v2_row_norms<<<Bz * Tz, 256, 0, stream>>>(q, f, qn, fn);
    w2v2_contrastive<<<Bz * Tz, 32, 0, stream>>>(q, f, negidx, mask, qn, fn, out);
}
